// T5Encoder_1597727834151
// MI455X (gfx1250) — compile-verified, matrix operand order checked by reading, not run on hardware
//
#include <hip/hip_runtime.h>
#include <hip/hip_bf16.h>
#include <math.h>

// ---------------------------------------------------------------------------
// T5 encoder for gfx1250 (MI455X). All GEMMs use v_wmma_f32_16x16x32_bf16.
// f32 -> bf16 conversion happens ONCE at LDS staging; LDS panels are stored
// pre-packed in the WMMA fragment layout so inner loops are ds_load_b128 +
// v_wmma only. Global->register panel loads are software-pipelined one K-step
// ahead so VMEM latency hides behind WMMA compute.
#define BB   4
#define SS   512
#define DD   768
#define HH   12
#define DHH  64
#define DFFN 3072
#define LL   6
#define NBUK 32
#define BS   (BB*SS)          // 2048 token rows
#define MASK_MIN -65504.0f
#define LDSTR 48              // padded panel stride (elements): 96B, 32B-aligned
// Workspace requirement: 6*BS*DD + BS*DFFN floats + SS*SS ints  ~= 64 MB.
// ---------------------------------------------------------------------------

typedef __attribute__((ext_vector_type(16))) __bf16 bf16x16;
typedef __attribute__((ext_vector_type(2)))  __bf16 bf16x2;
typedef __attribute__((ext_vector_type(8)))  float  f32x8;
typedef __attribute__((ext_vector_type(4)))  float  f32x4;

__device__ __forceinline__ __bf16 to_bf16(float f) { return (__bf16)f; }

__device__ __forceinline__ f32x8 wmma_bf16(bf16x16 a, bf16x16 b, f32x8 c) {
  // 8 args: (neg_a, A, neg_b, B, c_mod, C, reuse_a, reuse_b)
  return __builtin_amdgcn_wmma_f32_16x16x32_bf16(false, a, false, b,
                                                 (short)0, c, false, false);
}

// K index held by element e of a 16-bit A/B fragment (ISA 7.12.2, wave32):
// lanes 0-15 -> K base 0, lanes 16-31 -> K base 8; elems 0..7 cover base..base+7,
// elems 8..15 cover 16+base..16+base+7.
__device__ __forceinline__ int frag_k(int e, int kb) {
  return ((e >> 3) << 4) + kb + (e & 7);
}
// Inverse mapping: for k in [0,32), which (half, elem) holds it.
__device__ __forceinline__ int frag_e(int k)  { return ((k >> 4) << 3) + (k & 7); }
__device__ __forceinline__ int frag_h(int k)  { return (k >> 3) & 1; }

// Build a fragment whose 16 elements are the two contiguous 8-float runs
// base[kb..kb+7] and base[16+kb..16+kb+7]  ->  4x global_load_b128 + cvt.
__device__ __forceinline__ bf16x16 load_frag_rowmajor(const float* base, int kb) {
  const f32x4 v0 = *(const f32x4*)(base + kb);
  const f32x4 v1 = *(const f32x4*)(base + kb + 4);
  const f32x4 v2 = *(const f32x4*)(base + 16 + kb);
  const f32x4 v3 = *(const f32x4*)(base + 16 + kb + 4);
  bf16x16 f;
#pragma unroll
  for (int i = 0; i < 4; ++i) {
    f[i]      = to_bf16(v0[i]);
    f[4 + i]  = to_bf16(v1[i]);
    f[8 + i]  = to_bf16(v2[i]);
    f[12 + i] = to_bf16(v3[i]);
  }
  return f;
}

// ---------------------------------------------------------------------------
// Embedding gather: x[token, :] = embedding[input_ids[token], :]
// ---------------------------------------------------------------------------
__global__ __launch_bounds__(256)
void embed_kernel(const int* __restrict__ ids, const float* __restrict__ emb,
                  float* __restrict__ x) {
  const int token = blockIdx.x;
  const int id = ids[token];
  const float* src = emb + (size_t)id * DD;
  float* dst = x + (size_t)token * DD;
  for (int d = threadIdx.x; d < DD; d += 256) dst[d] = src[d];
}

// ext_mask[b,s] = (1 - attention_mask[b,s]) * MASK_MIN
__global__ void mask_kernel(const int* __restrict__ am, float* __restrict__ em, int n) {
  int i = blockIdx.x * 256 + threadIdx.x;
  if (i < n) em[i] = (1.0f - (float)am[i]) * MASK_MIN;
}

// T5 bidirectional relative position buckets, buckets[q*SS + k]
__global__ void bucket_kernel(int* __restrict__ buckets) {
  int idx = blockIdx.x * 256 + threadIdx.x;
  if (idx >= SS * SS) return;
  int q = idx >> 9, k = idx & 511;
  int rel = k - q;                       // memory - query
  int b = (rel > 0) ? 16 : 0;            // nb = NUM_BUCKETS/2 = 16
  int rp = (rel < 0) ? -rel : rel;
  int v;
  if (rp < 8) {                          // max_exact = 8
    v = rp;
  } else {
    int large = 8 + (int)(__logf((float)rp * 0.125f) * (8.0f / __logf(16.0f)));
    v = (large < 15) ? large : 15;
  }
  buckets[idx] = b + v;
}

// ---------------------------------------------------------------------------
// RMSNorm (T5 style): y = w * x * rsqrt(mean(x^2) + eps).  One block per row.
// ---------------------------------------------------------------------------
__global__ __launch_bounds__(256)
void rmsnorm_kernel(const float* __restrict__ X, const float* __restrict__ W,
                    float* __restrict__ Y) {
  __shared__ float red[8];
  const int row = blockIdx.x;
  const float* x = X + (size_t)row * DD;
  float s = 0.0f;
  for (int d = threadIdx.x; d < DD; d += 256) { float v = x[d]; s += v * v; }
  for (int off = 16; off > 0; off >>= 1) s += __shfl_xor(s, off, 32);
  if ((threadIdx.x & 31) == 0) red[threadIdx.x >> 5] = s;
  __syncthreads();
  if (threadIdx.x == 0) {
    float t = 0.0f;
    for (int i = 0; i < 8; ++i) t += red[i];
    red[0] = t;
  }
  __syncthreads();
  const float inv = rsqrtf(red[0] * (1.0f / DD) + 1e-6f);
  float* y = Y + (size_t)row * DD;
  for (int d = threadIdx.x; d < DD; d += 256) y[d] = W[d] * x[d] * inv;
}

// ---------------------------------------------------------------------------
// WMMA GEMM: C[M,N] = A[M,K] @ B[K,N]  (+residual, +relu via flags)
// Block = 256 threads (8 waves) -> 128x128 C tile.
// Wave w owns a 32x64 sub-tile -> 2 A frags x 4 B frags -> 8 WMMA / K-step.
// Software pipeline: global loads for panel k+32 are issued right after the
// barrier, WMMAs for panel k execute while those loads are in flight, and the
// register data is converted/stored to LDS at the top of the next iteration.
// flags: bit0 = relu, bit1 = add residual
// ---------------------------------------------------------------------------
struct StageRegs {
  float2 a[8];               // A: thread's 8 K-pairs (rows r, r+16, ...)
  float  b0[8], b1[8];       // B: thread's 8 K-pairs along K (stride 4)
};

__device__ __forceinline__ void load_panel(StageRegs& s,
                                           const float* __restrict__ A,
                                           const float* __restrict__ Bm,
                                           int rowBase, int colBase,
                                           int K, int N, int k0, int tid) {
  const float* gA = A + (size_t)(rowBase + (tid >> 4)) * K + k0 + (tid & 15) * 2;
#pragma unroll
  for (int i = 0; i < 8; ++i)
    s.a[i] = *(const float2*)(gA + (size_t)(i * 16) * K);
  const float* gB = Bm + (size_t)(k0 + (tid >> 7) * 2) * N + colBase + (tid & 127);
#pragma unroll
  for (int i = 0; i < 8; ++i) {
    s.b0[i] = gB[(size_t)(4 * i) * N];
    s.b1[i] = gB[(size_t)(4 * i + 1) * N];
  }
}

__device__ __forceinline__ void store_panel(const StageRegs& s,
                                            __bf16* pa, __bf16* pb, int tid) {
  const int r = tid >> 4, c = (tid & 15) * 2;       // A: row r+16i, even k = c
#pragma unroll
  for (int i = 0; i < 8; ++i) {
    bf16x2 t; t.x = to_bf16(s.a[i].x); t.y = to_bf16(s.a[i].y);
    *(bf16x2*)&pa[(r + i * 16) * LDSTR + frag_h(c) * 16 + frag_e(c)] = t;
  }
  const int kbase = (tid >> 7) * 2, cb = tid & 127; // B: col cb, even k = kbase+4i
#pragma unroll
  for (int i = 0; i < 8; ++i) {
    const int k = kbase + 4 * i;
    bf16x2 t; t.x = to_bf16(s.b0[i]); t.y = to_bf16(s.b1[i]);
    *(bf16x2*)&pb[cb * LDSTR + frag_h(k) * 16 + frag_e(k)] = t;
  }
}

__global__ __launch_bounds__(256)
void gemm_wmma_kernel(const float* __restrict__ A, const float* __restrict__ Bm,
                      const float* __restrict__ Res, float* __restrict__ C,
                      int M, int N, int K, int flags) {
  __shared__ __attribute__((aligned(32))) __bf16 pa[128 * LDSTR]; // 12 KB
  __shared__ __attribute__((aligned(32))) __bf16 pb[128 * LDSTR]; // 12 KB
  const int tid  = threadIdx.x;
  const int wave = tid >> 5;
  const int lane = tid & 31;
  const int hl   = lane & 15;
  const int half = lane >> 4;          // 0: K base 0, 1: K base 8
  const int rowBase = blockIdx.y * 128;
  const int colBase = blockIdx.x * 128;
  const int mw = (wave >> 1) * 32;     // wave row offset within block tile
  const int nw = (wave & 1) * 64;      // wave col offset within block tile

  f32x8 acc[2][4] = {};
  StageRegs sr;

  load_panel(sr, A, Bm, rowBase, colBase, K, N, 0, tid);

  for (int k0 = 0; k0 < K; k0 += 32) {
    store_panel(sr, pa, pb, tid);      // regs (panel k0) -> packed bf16 LDS
    __syncthreads();

    if (k0 + 32 < K)                   // issue next panel's loads now;
      load_panel(sr, A, Bm, rowBase, colBase, K, N, k0 + 32, tid);
    if (k0 + 64 < K)                   // and nudge the one after into L2
      __builtin_prefetch(A + (size_t)(rowBase + (tid >> 1)) * K + (k0 + 64), 0, 1);

    const bf16x16 a0 = *(const bf16x16*)&pa[(mw + hl)      * LDSTR + half * 16];
    const bf16x16 a1 = *(const bf16x16*)&pa[(mw + 16 + hl) * LDSTR + half * 16];
#pragma unroll
    for (int t = 0; t < 4; ++t) {
      const bf16x16 bf = *(const bf16x16*)&pb[(nw + t * 16 + hl) * LDSTR + half * 16];
      acc[0][t] = wmma_bf16(a0, bf, acc[0][t]);
      acc[1][t] = wmma_bf16(a1, bf, acc[1][t]);
    }
    __syncthreads();
  }

  // Epilogue: C/D layout — VGPR r holds M=r (lanes 0-15) / M=8+r (lanes 16-31)
#pragma unroll
  for (int m = 0; m < 2; ++m) {
#pragma unroll
    for (int t = 0; t < 4; ++t) {
#pragma unroll
      for (int r = 0; r < 8; ++r) {
        const int row = rowBase + mw + m * 16 + half * 8 + r;
        const int col = colBase + nw + t * 16 + hl;
        const size_t idx = (size_t)row * N + col;
        float v = acc[m][t][r];
        if (flags & 2) v += Res[idx];
        if (flags & 1) v = fmaxf(v, 0.0f);
        C[idx] = v;
      }
    }
  }
}

// ---------------------------------------------------------------------------
// Fused attention for one layer. Grid = (S/16 q-tiles, B*H). Block = 256 (8 waves).
// Phase 1: scores = Q Kt (WMMA, K=64) + rel bias + ext mask -> LDS [16 x 512]
//          (fragments fetched as 4x global_load_b128 each)
// Phase 2: per-row softmax (wave reductions); P -> d_out AND fragment-packed
//          bf16 LDS for phase 3.
// Phase 3: ctx = P V (WMMA, K=512); waves 0..3 own the 4 dh tiles; A-frags are
//          contiguous ds_load_b128 from the packed P array.
// ---------------------------------------------------------------------------
__global__ __launch_bounds__(256)
void attention_kernel(const float* __restrict__ Q, const float* __restrict__ Kb,
                      const float* __restrict__ V,
                      const float* __restrict__ relEmb,   // rel_emb + l*NBUK*HH
                      const int*   __restrict__ buckets,  // [S,S]
                      const float* __restrict__ extMask,  // [B,S]
                      float* __restrict__ Pout,           // attn out + l*B*H*S*S
                      float* __restrict__ Ctx) {
  __shared__ float sc[16 * SS];                                  // 32 KB
  __shared__ __attribute__((aligned(32))) __bf16 pp[16 * 16 * 32]; // 16 KB packed P
  const int tid  = threadIdx.x;
  const int wave = tid >> 5;
  const int lane = tid & 31;
  const int hl   = lane & 15;
  const int half = lane >> 4;
  const int kb   = half * 8;
  const int bh   = blockIdx.y;
  const int b    = bh / HH;
  const int h    = bh - b * HH;
  const int q0   = blockIdx.x * 16;
  const size_t tokBase = (size_t)b * SS;
  const int headOff = h * DHH;

  // ---- Phase 1: this wave computes score columns [wave*64, wave*64+64) ----
  f32x8 acc[4] = {};
#pragma unroll
  for (int k0 = 0; k0 < DHH; k0 += 32) {
    const bf16x16 afrag =
        load_frag_rowmajor(Q + (tokBase + q0 + hl) * DD + headOff + k0, kb);
#pragma unroll
    for (int t = 0; t < 4; ++t) {
      const int key = wave * 64 + t * 16 + hl;
      const bf16x16 bfrag =   // B[k=dh, n=key] = K[key, dh] (dh contiguous)
          load_frag_rowmajor(Kb + (tokBase + key) * DD + headOff + k0, kb);
      acc[t] = wmma_bf16(afrag, bfrag, acc[t]);
    }
  }
  // scores + relative-position bias + extended mask -> LDS
#pragma unroll
  for (int t = 0; t < 4; ++t) {
#pragma unroll
    for (int r = 0; r < 8; ++r) {
      const int row = kb + r;                  // q row within tile
      const int col = wave * 64 + t * 16 + hl; // key index
      const int qg  = q0 + row;
      const float bias = relEmb[buckets[qg * SS + col] * HH + h]
                       + extMask[b * SS + col];
      sc[row * SS + col] = acc[t][r] + bias;
    }
  }
  __syncthreads();

  // ---- Phase 2: softmax, wave w handles rows 2w and 2w+1 ----
  for (int rr = 0; rr < 2; ++rr) {
    const int row = wave * 2 + rr;
    float mx = -3.4e38f;
    for (int c = lane; c < SS; c += 32) mx = fmaxf(mx, sc[row * SS + c]);
    for (int off = 16; off > 0; off >>= 1) mx = fmaxf(mx, __shfl_xor(mx, off, 32));
    float sum = 0.0f;
    for (int c = lane; c < SS; c += 32) {
      float e = __expf(sc[row * SS + c] - mx);
      sc[row * SS + c] = e;
      sum += e;
    }
    for (int off = 16; off > 0; off >>= 1) sum += __shfl_xor(sum, off, 32);
    const float inv = 1.0f / sum;
    const size_t pbase = ((size_t)bh * SS + (q0 + row)) * SS;
    for (int c = lane; c < SS; c += 32) {
      const float p = sc[row * SS + c] * inv;
      Pout[pbase + c] = p;                       // attention probs output
      const int kt = c >> 5, kw = c & 31;        // pack P for phase-3 A frags
      pp[(kt * 16 + row) * 32 + frag_h(kw) * 16 + frag_e(kw)] = to_bf16(p);
    }
  }
  __syncthreads();

  // ---- Phase 3: ctx = P @ V ; waves 0..3 own dh tiles 0..3 ----
  if (wave < 4) {
    f32x8 cacc = {};
    for (int kt = 0; kt < SS / 32; ++kt) {
      const bf16x16 afrag = *(const bf16x16*)&pp[(kt * 16 + hl) * 32 + half * 16];
      bf16x16 bfrag;
#pragma unroll
      for (int e = 0; e < 16; ++e)   // B[k=key, n=dh] = V[key, dh]
        bfrag[e] = to_bf16(V[(tokBase + kt * 32 + frag_k(e, kb)) * DD
                             + headOff + wave * 16 + hl]);
      cacc = wmma_bf16(afrag, bfrag, cacc);
    }
#pragma unroll
    for (int r = 0; r < 8; ++r) {
      Ctx[(tokBase + q0 + kb + r) * DD + headOff + wave * 16 + hl] = cacc[r];
    }
  }
}

// ---------------------------------------------------------------------------
extern "C" void kernel_launch(void* const* d_in, const int* in_sizes, int n_in,
                              void* d_out, int out_size, void* d_ws, size_t ws_size,
                              hipStream_t stream) {
  const int*   input_ids  = (const int*)  d_in[0];
  const int*   attn_mask  = (const int*)  d_in[1];
  const float* embedding  = (const float*)d_in[2];
  const float* ln1_w      = (const float*)d_in[3];
  const float* q_w        = (const float*)d_in[4];
  const float* k_w        = (const float*)d_in[5];
  const float* v_w        = (const float*)d_in[6];
  const float* o_w        = (const float*)d_in[7];
  const float* rel_emb    = (const float*)d_in[8];
  const float* ln2_w      = (const float*)d_in[9];
  const float* ff1_w      = (const float*)d_in[10];
  const float* ff2_w      = (const float*)d_in[11];
  const float* final_ln_w = (const float*)d_in[12];

  // Outputs, concatenated flat in return order: out, attn_probs, ext_mask
  float* out_x    = (float*)d_out;                              // [B,S,D]
  float* out_p    = out_x + (size_t)BS * DD;                    // [L,B,H,S,S]
  float* out_mask = out_p + (size_t)LL * BB * HH * SS * SS;     // [B,1,1,S]

  // Workspace layout (f32 unless noted)
  float* x    = (float*)d_ws;            // [BS, D]   residual stream
  float* hbuf = x    + (size_t)BS * DD;  // [BS, D]   normed activations
  float* qb   = hbuf + (size_t)BS * DD;  // [BS, D]
  float* kbuf = qb   + (size_t)BS * DD;  // [BS, D]
  float* vbuf = kbuf + (size_t)BS * DD;  // [BS, D]
  float* ctx  = vbuf + (size_t)BS * DD;  // [BS, D]
  float* ffb  = ctx  + (size_t)BS * DD;  // [BS, DFF]
  int*   bkt  = (int*)(ffb + (size_t)BS * DFFN);   // [S, S]

  const dim3 blk(256);

  // Prologue
  embed_kernel<<<dim3(BS), blk, 0, stream>>>(input_ids, embedding, x);
  mask_kernel<<<dim3((BB * SS + 255) / 256), blk, 0, stream>>>(attn_mask, out_mask, BB * SS);
  bucket_kernel<<<dim3((SS * SS + 255) / 256), blk, 0, stream>>>(bkt);

  const dim3 gemmDD(DD / 128, BS / 128);     // (6, 16)
  const dim3 gemmFF1(DFFN / 128, BS / 128);  // (24, 16)
  const dim3 attnGrid(SS / 16, BB * HH);     // (32, 48)

  for (int l = 0; l < LL; ++l) {
    const float* qw  = q_w  + (size_t)l * DD * DD;
    const float* kw  = k_w  + (size_t)l * DD * DD;
    const float* vw  = v_w  + (size_t)l * DD * DD;
    const float* ow  = o_w  + (size_t)l * DD * DD;
    const float* f1w = ff1_w + (size_t)l * DD * DFFN;
    const float* f2w = ff2_w + (size_t)l * DFFN * DD;
    const float* re  = rel_emb + (size_t)l * NBUK * HH;
    float* pl = out_p + (size_t)l * BB * HH * SS * SS;

    // h = RMSNorm(x, ln1)
    rmsnorm_kernel<<<dim3(BS), blk, 0, stream>>>(x, ln1_w + (size_t)l * DD, hbuf);
    // q/k/v = h @ W
    gemm_wmma_kernel<<<gemmDD, blk, 0, stream>>>(hbuf, qw, hbuf, qb,   BS, DD, DD, 0);
    gemm_wmma_kernel<<<gemmDD, blk, 0, stream>>>(hbuf, kw, hbuf, kbuf, BS, DD, DD, 0);
    gemm_wmma_kernel<<<gemmDD, blk, 0, stream>>>(hbuf, vw, hbuf, vbuf, BS, DD, DD, 0);
    // attention (scores + bias + mask + softmax + P@V), P -> output
    attention_kernel<<<attnGrid, blk, 0, stream>>>(qb, kbuf, vbuf, re, bkt,
                                                   out_mask, pl, ctx);
    // x = x + ctx @ o_w   (residual add fused in epilogue)
    gemm_wmma_kernel<<<gemmDD, blk, 0, stream>>>(ctx, ow, x, x, BS, DD, DD, 2);
    // f = RMSNorm(x, ln2)
    rmsnorm_kernel<<<dim3(BS), blk, 0, stream>>>(x, ln2_w + (size_t)l * DD, hbuf);
    // ff = relu(f @ ff1_w)
    gemm_wmma_kernel<<<gemmFF1, blk, 0, stream>>>(hbuf, f1w, hbuf, ffb, BS, DFFN, DD, 1);
    // x = x + ff @ ff2_w
    gemm_wmma_kernel<<<gemmDD, blk, 0, stream>>>(ffb, f2w, x, x, BS, DD, DFFN, 2);
  }

  // out = RMSNorm(x, final_ln_w)
  rmsnorm_kernel<<<dim3(BS), blk, 0, stream>>>(x, final_ln_w, out_x);
}
